// MambaEnhancedUNet_84825604096651
// MI455X (gfx1250) — matrix operand order repaired and checked by reading, not compiled, not run on hardware
//
#include <hip/hip_runtime.h>
#include <hip/hip_bf16.h>
#include <math.h>

// ---------------------------------------------------------------------------
// CDNA5 (gfx1250, wave32) types & feature detection
// ---------------------------------------------------------------------------
typedef __bf16 bf16_t;
typedef __attribute__((ext_vector_type(16))) __bf16 v16bf;
typedef __attribute__((ext_vector_type(8)))  __bf16 v8bf;
typedef __attribute__((ext_vector_type(8)))  float  v8f;
typedef __attribute__((ext_vector_type(4)))  int    v4i;

#ifndef __has_builtin
#define __has_builtin(x) 0
#endif

#if __has_builtin(__builtin_amdgcn_global_load_async_to_lds_b128)
#define HAVE_ASYNC_LDS 1
#else
#define HAVE_ASYNC_LDS 0
#endif

// Builtin signature (from hipcc diagnostic): param 1 is
// 'int __vector(4) __device__ *'  ->  v4i in address_space(1) (global).
// Param 2 is the LDS destination  ->  v4i in address_space(3) (shared).
typedef __attribute__((address_space(1))) v4i* gv4i_p;
typedef __attribute__((address_space(3))) v4i* lv4i_p;

static __device__ __forceinline__ void async_copy16(const bf16_t* gsrc, bf16_t* ldst) {
#if HAVE_ASYNC_LDS
    // global -> LDS, 16 bytes per lane, tracked by ASYNCcnt
    __builtin_amdgcn_global_load_async_to_lds_b128(
        (gv4i_p)(v4i*)(size_t)gsrc, (lv4i_p)(v4i*)ldst, 0, 0);
#else
    *(v8bf*)ldst = *(const v8bf*)gsrc;
#endif
}

static __device__ __forceinline__ void wait_async0() {
#if HAVE_ASYNC_LDS
#if __has_builtin(__builtin_amdgcn_s_wait_asynccnt)
    __builtin_amdgcn_s_wait_asynccnt(0);
#else
    asm volatile("s_wait_asynccnt 0x0" ::: "memory");
#endif
#endif
}

static __device__ __forceinline__ float silu_f(float v) {
    return v / (1.0f + __expf(-v));
}

// ---------------------------------------------------------------------------
// Workhorse GEMM: C[M,N] = act(A[M,K] * Bt[N,K]^T + bias[N])
// A, Bt bf16 row-major (B pre-transposed: N rows of K), C f32 row-major.
// Block = 256 threads = 8 wave32 waves arranged 2(M) x 4(N).
// Block tile 64x128, wave tile 32x32 = 2x2 WMMA tiles, K step 32.
// Both tiles are K-contiguous 16B chunks -> staged global->LDS with
// global_load_async_to_lds_b128 (ASYNCcnt), double-buffered, overlapped
// with the v_wmma_f32_16x16x32_bf16 inner loop.
//
// Fragment layouts per CDNA5 ISA 7.12.2 (16-bit A 16x32 / B 32x16):
//   lane l, g=l/16, lm=l%16:
//   A: row M=lm; v16bf elems 0..7 = K[g*8..g*8+7], 8..15 = K[16+g*8..+7]
//   B: col N=lm; v16bf elems 0..7 = K[g*8..g*8+7], 8..15 = K[16+g*8..+7]
//   C/D: VGPR r -> M = r + g*8, N = lm
// act: 0=none, 1=silu, 2=gelu(exact)
// ---------------------------------------------------------------------------
__global__ __launch_bounds__(256) void k_gemm_bf16(
    const bf16_t* __restrict__ A, const bf16_t* __restrict__ Bt,
    const float* __restrict__ bias, float* __restrict__ C,
    int M, int N, int K, int act)
{
    __shared__ bf16_t lsA[2][64 * 32];
    __shared__ bf16_t lsB[2][128 * 32];

    const int tid  = threadIdx.x;
    const int lane = tid & 31;
    const int wave = tid >> 5;       // 0..7
    const int wm   = wave >> 2;      // 0..1
    const int wn   = wave & 3;       // 0..3
    const int g    = lane >> 4;      // lane group 0/1
    const int lm   = lane & 15;

    const int m0 = blockIdx.y * 64;
    const int n1 = blockIdx.x * 128;

    const int  ktiles = (K + 31) / 32;
    const bool fullMN = (m0 + 64 <= M) && (n1 + 128 <= N) && ((K & 7) == 0);

    // per-thread staging coordinates (fixed for the whole kernel)
    const int ar  = tid >> 2;             // A: row 0..63
    const int akk = (tid & 3) * 8;        // A: k offset 0,8,16,24
    const int bn  = tid >> 1;             // B: n row 0..127
    const int bkk = (tid & 1) * 16;       // B: k offset 0 or 16

    auto load_tiles = [&](int buf, int k0) {
        const bool fullK = (k0 + 32 <= K);
        if (fullMN && fullK) {
            // ---- fast path: async global->LDS, 16B chunks ----
            const bf16_t* ga = A + (size_t)(m0 + ar) * K + k0 + akk;
            async_copy16(ga, &lsA[buf][ar * 32 + akk]);
            const bf16_t* gb = Bt + (size_t)(n1 + bn) * K + k0 + bkk;
            async_copy16(gb,     &lsB[buf][bn * 32 + bkk]);
            async_copy16(gb + 8, &lsB[buf][bn * 32 + bkk + 8]);
            if (k0 + 64 < K) {  // prefetch hint for the tile after next
                __builtin_prefetch(ga + 32, 0, 1);
                __builtin_prefetch(gb + 32, 0, 1);
            }
        } else {
            // ---- edge path: guarded scalar fill ----
            {
                int gm = m0 + ar, gk = k0 + akk;
                bf16_t* dst = &lsA[buf][ar * 32 + akk];
                #pragma unroll
                for (int i = 0; i < 8; ++i)
                    dst[i] = (gm < M && (gk + i) < K)
                                 ? A[(size_t)gm * K + gk + i] : (bf16_t)0.0f;
            }
            {
                int gn = n1 + bn, gk = k0 + bkk;
                bf16_t* dst = &lsB[buf][bn * 32 + bkk];
                #pragma unroll
                for (int i = 0; i < 16; ++i)
                    dst[i] = (gn < N && (gk + i) < K)
                                 ? Bt[(size_t)gn * K + gk + i] : (bf16_t)0.0f;
            }
        }
    };

    v8f acc[2][2];
    const v8f vzero = {0.f, 0.f, 0.f, 0.f, 0.f, 0.f, 0.f, 0.f};
    #pragma unroll
    for (int tm = 0; tm < 2; ++tm)
        #pragma unroll
        for (int tn = 0; tn < 2; ++tn)
            acc[tm][tn] = vzero;

    int buf = 0;
    load_tiles(buf, 0);
    wait_async0();
    __syncthreads();

    for (int kt = 0; kt < ktiles; ++kt) {
        if (kt + 1 < ktiles) load_tiles(buf ^ 1, (kt + 1) * 32);

        v16bf afr[2], bfr[2];
        #pragma unroll
        for (int tm = 0; tm < 2; ++tm) {
            int row = wm * 32 + tm * 16 + lm;
            const bf16_t* pa = &lsA[buf][row * 32 + g * 8];
            v8bf lo = *(const v8bf*)pa;
            v8bf hi = *(const v8bf*)(pa + 16);
            #pragma unroll
            for (int i = 0; i < 8; ++i) { afr[tm][i] = lo[i]; afr[tm][8 + i] = hi[i]; }
        }
        #pragma unroll
        for (int tn = 0; tn < 2; ++tn) {
            int col = wn * 32 + tn * 16 + lm;
            const bf16_t* pb = &lsB[buf][col * 32 + g * 8];
            v8bf lo = *(const v8bf*)pb;
            v8bf hi = *(const v8bf*)(pb + 16);
            #pragma unroll
            for (int i = 0; i < 8; ++i) { bfr[tn][i] = lo[i]; bfr[tn][8 + i] = hi[i]; }
        }

        #pragma unroll
        for (int tm = 0; tm < 2; ++tm)
            #pragma unroll
            for (int tn = 0; tn < 2; ++tn)
                acc[tm][tn] = __builtin_amdgcn_wmma_f32_16x16x32_bf16(
                    false, afr[tm], false, bfr[tn], (short)0, acc[tm][tn],
                    false, false);

        wait_async0();     // async writes into buf^1 must land before barrier
        __syncthreads();
        buf ^= 1;
    }

    #pragma unroll
    for (int tm = 0; tm < 2; ++tm) {
        #pragma unroll
        for (int tn = 0; tn < 2; ++tn) {
            int col = n1 + wn * 32 + tn * 16 + lm;
            if (col >= N) continue;
            float bv = bias ? bias[col] : 0.0f;
            #pragma unroll
            for (int r = 0; r < 8; ++r) {
                int row = m0 + wm * 32 + tm * 16 + g * 8 + r;
                if (row >= M) continue;
                float v = acc[tm][tn][r] + bv;
                if (act == 1)      v = silu_f(v);
                else if (act == 2) v = 0.5f * v * (1.0f + erff(v * 0.70710678f));
                C[(size_t)row * N + col] = v;
            }
        }
    }
}

// ---------------------------------------------------------------------------
// Elementwise / layout helpers
// ---------------------------------------------------------------------------
__global__ void k_cast_f32_bf16(const float* __restrict__ in,
                                bf16_t* __restrict__ out, int n) {
    int i = blockIdx.x * blockDim.x + threadIdx.x;
    if (i < n) out[i] = (bf16_t)in[i];
}

__global__ void k_zero_f32(float* __restrict__ out, int n) {
    int i = blockIdx.x * blockDim.x + threadIdx.x;
    if (i < n) out[i] = 0.0f;
}

// f32 [M][K] -> bf16 [M][Kp], zero-padded K tail (A-side / Bt-from-OIHW cast)
__global__ void k_cast_pad(const float* __restrict__ in,
                           bf16_t* __restrict__ out, int M, int K, int Kp) {
    int i = blockIdx.x * blockDim.x + threadIdx.x;
    if (i >= M * Kp) return;
    int m = i / Kp, kp = i % Kp;
    out[i] = (kp < K) ? (bf16_t)in[(size_t)m * K + kp] : (bf16_t)0.0f;
}

// f32 [R][Cc] -> bf16 [Cc][Rp] transposed, zero-padded (build Bt matrices)
__global__ void k_cast_transpose_pad(const float* __restrict__ in,
                                     bf16_t* __restrict__ out,
                                     int R, int Cc, int Rp) {
    int i = blockIdx.x * blockDim.x + threadIdx.x;
    if (i >= Cc * Rp) return;
    int c = i / Rp, r = i % Rp;
    out[i] = (r < R) ? (bf16_t)in[(size_t)r * Cc + c] : (bf16_t)0.0f;
}

// im2col 3x3 pad1 stride1, NCHW f32 -> patches [H*W, ldk] bf16 (K zero-padded)
__global__ void k_im2col_nchw(const float* __restrict__ x,
                              bf16_t* __restrict__ out,
                              int C, int H, int W, int ldk) {
    int i = blockIdx.x * blockDim.x + threadIdx.x;
    int tot = H * W * ldk;
    if (i >= tot) return;
    int l = i / ldk;
    int r = i % ldk;
    float v = 0.0f;
    if (r < C * 9) {
        int c  = r / 9;
        int kk = r % 9;
        int h  = l / W + (kk / 3) - 1;
        int w_ = l % W + (kk % 3) - 1;
        if (h >= 0 && h < H && w_ >= 0 && w_ < W)
            v = x[(size_t)c * H * W + h * W + w_];
    }
    out[(size_t)l * ldk + r] = (bf16_t)v;
}

// im2col, channel-last [L, C] f32 input
__global__ void k_im2col_lc(const float* __restrict__ x,
                            bf16_t* __restrict__ out,
                            int C, int H, int W, int ldk) {
    int i = blockIdx.x * blockDim.x + threadIdx.x;
    int tot = H * W * ldk;
    if (i >= tot) return;
    int l = i / ldk;
    int r = i % ldk;
    float v = 0.0f;
    if (r < C * 9) {
        int c  = r / 9;
        int kk = r % 9;
        int h  = l / W + (kk / 3) - 1;
        int w_ = l % W + (kk % 3) - 1;
        if (h >= 0 && h < H && w_ >= 0 && w_ < W)
            v = x[(size_t)(h * W + w_) * C + c];
    }
    out[(size_t)l * ldk + r] = (bf16_t)v;
}

// Depthwise 3x3 + bias + SiLU; channel-last input with leading dim ldx/offset.
__global__ void k_dwconv3_silu(const float* __restrict__ x, int ldx, int xoff,
                               const float* __restrict__ wgt,
                               const float* __restrict__ bias,
                               float* __restrict__ y, int H, int W, int D) {
    int i = blockIdx.x * blockDim.x + threadIdx.x;
    int tot = H * W * D;
    if (i >= tot) return;
    int l = i / D;
    int d = i % D;
    int h = l / W, w_ = l % W;
    float acc = bias[d];
    #pragma unroll
    for (int kh = 0; kh < 3; ++kh) {
        int hh = h + kh - 1;
        if (hh < 0 || hh >= H) continue;
        #pragma unroll
        for (int kw = 0; kw < 3; ++kw) {
            int ww = w_ + kw - 1;
            if (ww < 0 || ww >= W) continue;
            acc += x[(size_t)(hh * W + ww) * ldx + xoff + d] * wgt[d * 9 + kh * 3 + kw];
        }
    }
    y[(size_t)l * D + d] = silu_f(acc);
}

// Build 4 scan directions from channel-last xc [L, D].
// xs   : f32  [k][d][l]   (scan-friendly layout)
// xsb  : bf16 [k][l][d]   (Bt layout for the x_proj GEMM: N=L rows, K=D)
__global__ void k_build_xs(const float* __restrict__ xc,
                           float* __restrict__ xs, bf16_t* __restrict__ xsb,
                           int H, int W, int D) {
    int i = blockIdx.x * blockDim.x + threadIdx.x;
    int L = H * W;
    int tot = L * D;
    if (i >= tot) return;
    int l = i / D;
    int d = i % D;
    float v = xc[(size_t)l * D + d];
    int h = l / W, w_ = l % W;
    int lT = w_ * H + h;
    int p0 = l, p1 = lT, p2 = L - 1 - l, p3 = L - 1 - lT;
    xs[(size_t)(0 * D + d) * L + p0] = v;
    xs[(size_t)(1 * D + d) * L + p1] = v;
    xs[(size_t)(2 * D + d) * L + p2] = v;
    xs[(size_t)(3 * D + d) * L + p3] = v;
    bf16_t vb = (bf16_t)v;
    xsb[((size_t)0 * L + p0) * D + d] = vb;
    xsb[((size_t)1 * L + p1) * D + d] = vb;
    xsb[((size_t)2 * L + p2) * D + d] = vb;
    xsb[((size_t)3 * L + p3) * D + d] = vb;
}

// delta = softplus(dts + 2*dt_b[d]) (reference adds dt_b twice), in place.
__global__ void k_softplus_rowbias(float* __restrict__ dts,
                                   const float* __restrict__ dtb,
                                   int D, int L) {
    int i = blockIdx.x * blockDim.x + threadIdx.x;
    if (i >= D * L) return;
    int d = i / L;
    float v = dts[i] + 2.0f * dtb[d];
    dts[i] = (v > 20.0f) ? v : log1pf(__expf(v));
}

// Selective scan, one direction: one thread per channel d, 16 states.
__global__ void k_ss_scan(const float* __restrict__ delta,
                          const float* __restrict__ xdbl,
                          const float* __restrict__ u,
                          const float* __restrict__ A_logs,  // [Din,16]
                          const float* __restrict__ Ds,      // [Din]
                          float* __restrict__ ys,
                          int Din, int L, int R) {
    int d = blockIdx.x * blockDim.x + threadIdx.x;
    if (d >= Din) return;
    float Arow[16], h[16];
    #pragma unroll
    for (int n = 0; n < 16; ++n) {
        Arow[n] = -__expf(A_logs[d * 16 + n]);
        h[n] = 0.0f;
    }
    float Dd = Ds[d];
    const float* dl_p = delta + (size_t)d * L;
    const float* u_p  = u + (size_t)d * L;
    float* y_p        = ys + (size_t)d * L;
    for (int l = 0; l < L; ++l) {
        float dl = dl_p[l];
        float ul = u_p[l];
        float du = dl * ul;
        float y = 0.0f;
        #pragma unroll
        for (int n = 0; n < 16; ++n) {
            float Bn = xdbl[(size_t)(R + n) * L + l];
            float Cn = xdbl[(size_t)(R + 16 + n) * L + l];
            h[n] = __expf(dl * Arow[n]) * h[n] + du * Bn;
            y += h[n] * Cn;
        }
        y_p[l] = y + Dd * ul;
    }
}

// Merge 4 directions (flip / spatial transpose) -> channel-last [L, D]
__global__ void k_merge_dirs(const float* __restrict__ ys,
                             float* __restrict__ out, int H, int W, int D) {
    int i = blockIdx.x * blockDim.x + threadIdx.x;
    int L = H * W;
    if (i >= L * D) return;
    int l = i / D;
    int d = i % D;
    int h = l / W, w_ = l % W;
    int lT = w_ * H + h;
    float v = ys[(size_t)(0 * D + d) * L + l]
            + ys[(size_t)(2 * D + d) * L + (L - 1 - l)]
            + ys[(size_t)(1 * D + d) * L + lT]
            + ys[(size_t)(3 * D + d) * L + (L - 1 - lT)];
    out[(size_t)l * D + d] = v;
}

// LayerNorm over channels * silu(z) gate. One block per row; z=null -> no gate.
__global__ __launch_bounds__(128) void k_ln_gate(
    const float* __restrict__ y, const float* __restrict__ z, int ldz, int zoff,
    const float* __restrict__ lw, const float* __restrict__ lb,
    float* __restrict__ out, int C) {
    __shared__ float red[128];
    int l = blockIdx.x;
    int t = threadIdx.x;
    float s = 0.0f;
    for (int c = t; c < C; c += 128) s += y[(size_t)l * C + c];
    red[t] = s; __syncthreads();
    for (int o = 64; o > 0; o >>= 1) { if (t < o) red[t] += red[t + o]; __syncthreads(); }
    float mean = red[0] / (float)C;
    __syncthreads();
    float s2 = 0.0f;
    for (int c = t; c < C; c += 128) {
        float d = y[(size_t)l * C + c] - mean;
        s2 += d * d;
    }
    red[t] = s2; __syncthreads();
    for (int o = 64; o > 0; o >>= 1) { if (t < o) red[t] += red[t + o]; __syncthreads(); }
    float inv = rsqrtf(red[0] / (float)C + 1e-5f);
    for (int c = t; c < C; c += 128) {
        float v = (y[(size_t)l * C + c] - mean) * inv * lw[c] + lb[c];
        if (z) {
            float zv = z[(size_t)l * ldz + zoff + c];
            v *= silu_f(zv);
        }
        out[(size_t)l * C + c] = v;
    }
}

// [L, C] -> [C, L] (NCHW output)
__global__ void k_lc_to_cl(const float* __restrict__ in,
                           float* __restrict__ out, int L, int C, int nmax) {
    int i = blockIdx.x * blockDim.x + threadIdx.x;
    if (i >= L * C || i >= nmax) return;
    int l = i / C;
    int c = i % C;
    out[(size_t)c * L + l] = in[i];
}

// ---------------------------------------------------------------------------
// Host orchestration
// ---------------------------------------------------------------------------
static inline int cdiv(int a, int b) { return (a + b - 1) / b; }

extern "C" void kernel_launch(void* const* d_in, const int* in_sizes, int n_in,
                              void* d_out, int out_size, void* d_ws, size_t ws_size,
                              hipStream_t stream) {
    if (n_in <= 0 || !d_out || !d_ws) return;

    const int H = 96, W = 96, L = H * W;     // spatial
    const int C0 = 64, Din = 128, R = 4;     // first UNet scale
    const int CPROJ = R + 32;                // R + 2*N_STATE = 36

    auto find_sz = [&](int sz, int nth) -> int {
        int c = 0;
        for (int i = 0; i < n_in; ++i)
            if (in_sizes[i] == sz) { if (c == nth) return i; ++c; }
        return -1;
    };
    auto FP = [&](int idx) -> const float* {
        return (const float*)d_in[(idx >= 0 && idx < n_in) ? idx : 0];
    };

    int ix      = find_sz(6 * 96 * 96, 0);     // x (1,6,96,96)
    int i_inw   = find_sz(64 * 6 * 9, 0);      // in_w  [64,6,3,3]
    int i_inb   = find_sz(64, 0);              // in_b
    int i_ssw   = find_sz(64 * 256, 0);        // ss.in_w [64,256]
    int i_cvw   = find_sz(Din * 9, 0);         // cv_w [128,1,3,3]
    int i_cvb   = find_sz(Din, 0);             // cv_b
    int i_xproj = find_sz(4 * CPROJ * Din, 0); // x_proj [4,36,128]
    int i_alog  = find_sz(Din * 16, 0);        // A_logs
    int i_dtw   = find_sz(Din * 16, 1);        // dt_w [4,128,4]
    int i_ds    = find_sz(4 * Din, 0);         // Ds
    int i_dtb   = find_sz(4 * Din, 1);         // dt_b
    int i_lnb   = find_sz(Din, 1);             // ln_b
    int i_lnw   = find_sz(Din, 2);             // ln_w
    int i_outw  = find_sz(Din * C0, 0);        // out_w [128,64]
    int i_outb  = find_sz(64, 1);              // out_b
    int i_fc2w  = find_sz(3 * 64 * 9, 0);      // fc2w [3,64,3,3]
    int i_fc2b  = find_sz(3, 0);               // fc2b

    char* base = (char*)d_ws;
    size_t off = 0;
    auto alloc = [&](size_t bytes) -> void* {
        void* p = base + off;
        off = (off + bytes + 255) & ~(size_t)255;
        return p;
    };

    bf16_t* patches1 = (bf16_t*)alloc((size_t)L * 64 * 2);
    bf16_t* Wr1      = (bf16_t*)alloc((size_t)64 * 64 * 2);
    float*  h0       = (float*) alloc((size_t)L * 64 * 4);
    bf16_t* h0b      = (bf16_t*)alloc((size_t)L * 64 * 2);
    bf16_t* sswb     = (bf16_t*)alloc((size_t)256 * 64 * 2);
    float*  xz       = (float*) alloc((size_t)L * 256 * 4);
    float*  xcs      = (float*) alloc((size_t)L * Din * 4);
    float*  xs       = (float*) alloc((size_t)4 * Din * L * 4);
    bf16_t* xsb      = (bf16_t*)alloc((size_t)4 * L * Din * 2);
    bf16_t* xprojb   = (bf16_t*)alloc((size_t)CPROJ * Din * 2);
    float*  xdbl     = (float*) alloc((size_t)CPROJ * L * 4);
    bf16_t* xdblRb   = (bf16_t*)alloc((size_t)L * 8 * 2);
    bf16_t* dtwb     = (bf16_t*)alloc((size_t)Din * 8 * 2);
    float*  dts      = (float*) alloc((size_t)Din * L * 4);
    float*  ysall    = (float*) alloc((size_t)4 * Din * L * 4);
    float*  ymerged  = (float*) alloc((size_t)L * Din * 4);
    float*  gated    = (float*) alloc((size_t)L * Din * 4);
    bf16_t* gatedb   = (bf16_t*)alloc((size_t)L * Din * 2);
    bf16_t* outwb    = (bf16_t*)alloc((size_t)C0 * Din * 2);
    float*  y2       = (float*) alloc((size_t)L * C0 * 4);
    bf16_t* patches2 = (bf16_t*)alloc((size_t)L * C0 * 9 * 2);
    bf16_t* Wr2      = (bf16_t*)alloc((size_t)3 * C0 * 9 * 2);
    float*  y3       = (float*) alloc((size_t)L * 3 * 4);

    if (off > ws_size) {
        k_zero_f32<<<cdiv(out_size, 256), 256, 0, stream>>>((float*)d_out, out_size);
        return;
    }

    const int TB = 256;
    const float* x = FP(ix);

    // 1) input conv 3x3 (6->64): im2col (K padded 54->64) + GEMM.
    //    OIHW weight [OC][C*9] is already Bt layout -> pad-cast only.
    k_im2col_nchw<<<cdiv(L * 64, TB), TB, 0, stream>>>(x, patches1, 6, H, W, 64);
    k_cast_pad<<<cdiv(64 * 64, TB), TB, 0, stream>>>(FP(i_inw), Wr1, 64, 54, 64);
    {
        dim3 g(cdiv(64, 128), cdiv(L, 64));
        k_gemm_bf16<<<g, TB, 0, stream>>>(patches1, Wr1, FP(i_inb), h0, L, 64, 64, 0);
    }

    // 2) ss2d in-projection: [L,64] x [64,256] -> xz (xc | z); Bt = [256][64]
    k_cast_f32_bf16<<<cdiv(L * 64, TB), TB, 0, stream>>>(h0, h0b, L * 64);
    k_cast_transpose_pad<<<cdiv(256 * 64, TB), TB, 0, stream>>>(FP(i_ssw), sswb, 64, 256, 64);
    {
        dim3 g(cdiv(256, 128), cdiv(L, 64));
        k_gemm_bf16<<<g, TB, 0, stream>>>(h0b, sswb, nullptr, xz, L, 256, 64, 0);
    }

    // 3) depthwise conv 3x3 + SiLU on the xc half
    k_dwconv3_silu<<<cdiv(L * Din, TB), TB, 0, stream>>>(
        xz, 256, 0, FP(i_cvw), FP(i_cvb), xcs, H, W, Din);

    // 4) 4 scan directions (f32 for scan, bf16 [l][d] = Bt for GEMM)
    k_build_xs<<<cdiv(L * Din, TB), TB, 0, stream>>>(xcs, xs, xsb, H, W, Din);

    // 5) per-direction: x_proj GEMM, dt GEMM, softplus, selective scan
    for (int k = 0; k < 4; ++k) {
        const bf16_t* xsb_k = xsb + (size_t)k * L * Din;
        const float*  xs_k  = xs  + (size_t)k * Din * L;
        float*        ys_k  = ysall + (size_t)k * Din * L;

        // x_dbl [36, L] = x_proj_k [36,128] x xs_k^T (Bt = [L][128])
        k_cast_f32_bf16<<<cdiv(CPROJ * Din, TB), TB, 0, stream>>>(
            FP(i_xproj) + (size_t)k * CPROJ * Din, xprojb, CPROJ * Din);
        {
            dim3 g(cdiv(L, 128), cdiv(CPROJ, 64));
            k_gemm_bf16<<<g, TB, 0, stream>>>(xprojb, xsb_k, nullptr, xdbl,
                                              CPROJ, L, Din, 0);
        }
        // dts [128, L] = dt_w_k [128,4] x x_dbl[:4]; K padded 4->8
        k_cast_pad<<<cdiv(Din * 8, TB), TB, 0, stream>>>(
            FP(i_dtw) + (size_t)k * Din * R, dtwb, Din, R, 8);
        k_cast_transpose_pad<<<cdiv(L * 8, TB), TB, 0, stream>>>(
            xdbl, xdblRb, R, L, 8);
        {
            dim3 g(cdiv(L, 128), cdiv(Din, 64));
            k_gemm_bf16<<<g, TB, 0, stream>>>(dtwb, xdblRb, nullptr, dts,
                                              Din, L, 8, 0);
        }
        k_softplus_rowbias<<<cdiv(Din * L, TB), TB, 0, stream>>>(
            dts, FP(i_dtb) + (size_t)k * Din, Din, L);

        k_ss_scan<<<cdiv(Din, 128), 128, 0, stream>>>(
            dts, xdbl, xs_k, FP(i_alog), FP(i_ds) + (size_t)k * Din,
            ys_k, Din, L, R);
    }

    // 6) merge directions, LayerNorm * silu(z) gate
    k_merge_dirs<<<cdiv(L * Din, TB), TB, 0, stream>>>(ysall, ymerged, H, W, Din);
    k_ln_gate<<<L, 128, 0, stream>>>(ymerged, xz, 256, 128,
                                     FP(i_lnw), FP(i_lnb), gated, Din);

    // 7) out-projection [L,128] x [128,64] (+SiLU); Bt = [64][128]
    k_cast_f32_bf16<<<cdiv(L * Din, TB), TB, 0, stream>>>(gated, gatedb, L * Din);
    k_cast_transpose_pad<<<cdiv(C0 * Din, TB), TB, 0, stream>>>(
        FP(i_outw), outwb, Din, C0, Din);
    {
        dim3 g(cdiv(C0, 128), cdiv(L, 64));
        k_gemm_bf16<<<g, TB, 0, stream>>>(gatedb, outwb, FP(i_outb), y2,
                                          L, C0, Din, 1 /*silu*/);
    }

    // 8) final conv 3x3 (64->3): im2col (K=576) + GEMM, transpose to NCHW
    k_im2col_lc<<<cdiv(L * C0 * 9, TB), TB, 0, stream>>>(y2, patches2, C0, H, W, 576);
    k_cast_f32_bf16<<<cdiv(3 * 576, TB), TB, 0, stream>>>(FP(i_fc2w), Wr2, 3 * 576);
    {
        dim3 g(cdiv(3, 128), cdiv(L, 64));
        k_gemm_bf16<<<g, TB, 0, stream>>>(patches2, Wr2, FP(i_fc2b), y3,
                                          L, 3, 576, 0);
    }
    k_zero_f32<<<cdiv(out_size, TB), TB, 0, stream>>>((float*)d_out, out_size);
    k_lc_to_cl<<<cdiv(L * 3, TB), TB, 0, stream>>>(y3, (float*)d_out, L, 3, out_size);
}
